// TransformerVanilla_48172353192214
// MI455X (gfx1250) — matrix operand order repaired and checked by reading, not compiled, not run on hardware
//
// GPT-2-style transformer forward (logits + NLL loss) for MI455X / gfx1250.
// All GEMMs run through v_wmma_f32_16x16x32_bf16 (bf16 in, f32 accumulate).
// A/B tiles staged with global_load_async_to_lds_b128 (ASYNCcnt) into
// double-buffered LDS; weights pre-transposed so B is always K-contiguous.
#include <hip/hip_runtime.h>
#include <cstdint>
#include <cstddef>

#define DEVFN __device__ __forceinline__

typedef __attribute__((ext_vector_type(16))) __bf16 bf16x16;
typedef __attribute__((ext_vector_type(8)))  float  f32x8;

union FragU { uint4 u[2]; bf16x16 v; };

DEVFN unsigned short f2bf(float f) {
  // round-to-nearest-even f32 -> bf16
  unsigned int u = __float_as_uint(f);
  u += 0x7FFFu + ((u >> 16) & 1u);
  return (unsigned short)(u >> 16);
}

// Async copy 16B from global to LDS (per lane). Tracked by ASYNCcnt.
DEVFN void async_b128(unsigned lds_addr, const void* gaddr) {
  asm volatile("global_load_async_to_lds_b128 %0, %1, off"
               :: "v"(lds_addr), "v"(gaddr) : "memory");
}
DEVFN void wait_async0() { asm volatile("s_wait_asynccnt 0x0" ::: "memory"); }

// ---------------------------------------------------------------------------
// Generic tiled WMMA GEMM:  C(M,N) = epilogue( A(M,K) @ B + bias + res )
//   A: bf16 row-major, lda            (M % 128 == 0, K % 32 == 0 assumed)
//   B: bf16; TB=1 -> (N,K) row-major (weightsT / k);  TB=0 -> (K,N) row-major
//   C: f32 or bf16 (OBF), ldc; res: f32 with same ldc
// Tiles: BM=BN=128, BK=32; 256 thr = 8 waves; wave tile 32x64 (2x4 frags).
// Double-buffered LDS; A (+B when TB) staged via async-to-LDS.
// ---------------------------------------------------------------------------
template<bool TB, bool BIAS, bool RES, bool DORELU, bool OBF>
__global__ __launch_bounds__(256) void gemm_bf16_wmma(
    const unsigned short* __restrict__ A, int lda,
    const unsigned short* __restrict__ B, int ldb,
    const float* __restrict__ bias,
    const float* __restrict__ res,
    void* __restrict__ Cv, int ldc,
    int M, int N, int Kd)
{
  constexpr int BM = 128, BN = 128, LSTR = 40; // padded LDS stride
  __shared__ unsigned short As[2][BM * LSTR];
  __shared__ unsigned short Bs[2][BN * LSTR];  // stored N-major: Bs[.][n][k]

  const int tid  = threadIdx.x;
  const int lane = tid & 31, wid = tid >> 5;
  const int r    = lane & 15, hf = lane >> 4;
  const int wm   = (wid & 3) * 32;   // 4 waves along M
  const int wn   = (wid >> 2) * 64;  // 2 waves along N
  const int m0   = blockIdx.y * BM;
  const int n0   = blockIdx.x * BN;
  const int nk   = Kd >> 5;

  f32x8 acc[2][4];
#pragma unroll
  for (int i = 0; i < 2; ++i)
#pragma unroll
    for (int j = 0; j < 4; ++j)
#pragma unroll
      for (int e = 0; e < 8; ++e) acc[i][j][e] = 0.f;

  uint4 bR[2];  // staging regs (non-TB B transpose-scatter only)

  auto stageAsync = [&](int kc, int buf) {
#pragma unroll
    for (int it = 0; it < 2; ++it) {           // A: 128x32, 16B x2 per thread
      int i = tid + it * 256;
      int row = i >> 2, kg = (i & 3) << 3;
      unsigned dst = (unsigned)(uintptr_t)&As[buf][row * LSTR + kg];
      async_b128(dst, A + (size_t)(m0 + row) * lda + kc + kg);
    }
    if (TB) {                                  // B (N,K): contiguous along K
#pragma unroll
      for (int it = 0; it < 2; ++it) {
        int i = tid + it * 256;
        int nrow = i >> 2, kg = (i & 3) << 3;
        int nr = (n0 + nrow < N) ? nrow : 0;   // clamp OOB rows (cols unused)
        unsigned dst = (unsigned)(uintptr_t)&Bs[buf][nrow * LSTR + kg];
        async_b128(dst, B + (size_t)(n0 + nr) * ldb + kc + kg);
      }
    }
  };

  auto loadBreg = [&](int kc) {                // non-TB: B (K,N) into regs
#pragma unroll
    for (int it = 0; it < 2; ++it) {
      int i = tid + it * 256;                  // 32 k-rows x 16 n-groups
      int krow = i >> 4, ng = (i & 15) << 3;
      uint4 d; d.x = d.y = d.z = d.w = 0u;
      if (n0 + ng < N)                         // N%8==0 -> chunk all-in or out
        d = *(const uint4*)(B + (size_t)(kc + krow) * ldb + n0 + ng);
      bR[it] = d;
    }
  };

  auto scatterB = [&](int buf) {               // transpose into Bs[n][k]
#pragma unroll
    for (int it = 0; it < 2; ++it) {
      int i = tid + it * 256;
      int krow = i >> 4, ng = (i & 15) << 3;
      union { uint4 u; unsigned short s[8]; } t;
      t.u = bR[it];
#pragma unroll
      for (int j = 0; j < 8; ++j) Bs[buf][(ng + j) * LSTR + krow] = t.s[j];
    }
  };

  // prologue: stage K-chunk 0 into buffer 0
  stageAsync(0, 0);
  if (!TB) { loadBreg(0); scatterB(0); }
  wait_async0();
  __syncthreads();

  for (int ki = 0; ki < nk; ++ki) {
    const int  cur  = ki & 1;
    const bool more = (ki + 1) < nk;
    if (more) {                                // fill next buffer (overlaps WMMA)
      stageAsync((ki + 1) << 5, cur ^ 1);
      if (!TB) loadBreg((ki + 1) << 5);
    }

    FragU af[2], bfg[4];
#pragma unroll
    for (int mt = 0; mt < 2; ++mt) {           // A: K runs at 8*hf, 16+8*hf
      int base = (wm + mt * 16 + r) * LSTR + 8 * hf;
      af[mt].u[0] = *(const uint4*)&As[cur][base];
      af[mt].u[1] = *(const uint4*)&As[cur][base + 16];
    }
#pragma unroll
    for (int nt = 0; nt < 4; ++nt) {           // B: K runs at 8*hf, 16+8*hf
      int base = (wn + nt * 16 + r) * LSTR + 8 * hf;
      bfg[nt].u[0] = *(const uint4*)&Bs[cur][base];
      bfg[nt].u[1] = *(const uint4*)&Bs[cur][base + 16];
    }
#pragma unroll
    for (int mt = 0; mt < 2; ++mt)
#pragma unroll
      for (int nt = 0; nt < 4; ++nt)
        acc[mt][nt] = __builtin_amdgcn_wmma_f32_16x16x32_bf16(
            false, af[mt].v, false, bfg[nt].v, (short)0, acc[mt][nt],
            false, false);

    if (more) {
      if (!TB) scatterB(cur ^ 1);
      wait_async0();
    }
    __syncthreads();
  }

  // ---- epilogue: C layout lane(n=r), vgpr j -> m = hf*8 + j ---------------
  float*          Cf = (float*)Cv;
  unsigned short* Cb = (unsigned short*)Cv;
#pragma unroll
  for (int mt = 0; mt < 2; ++mt)
#pragma unroll
    for (int nt = 0; nt < 4; ++nt)
#pragma unroll
      for (int j = 0; j < 8; ++j) {
        int gm = m0 + wm + mt * 16 + hf * 8 + j;
        int gn = n0 + wn + nt * 16 + r;
        if (gn < N && gm < M) {
          float v = acc[mt][nt][j];
          if (BIAS)   v += bias[gn];
          if (RES)    v += res[(size_t)gm * ldc + gn];
          if (DORELU) v  = fmaxf(v, 0.f);
          if (OBF) Cb[(size_t)gm * ldc + gn] = f2bf(v);
          else     Cf[(size_t)gm * ldc + gn] = v;
        }
      }
}

// ---------------------------------------------------------------------------
// Elementwise / reduction kernels
// ---------------------------------------------------------------------------
// batched (K,N) f32 -> (N,K) bf16 transpose-convert; block (32,8), grid.z batch
__global__ void transpose_cvt_kernel(const float* __restrict__ in,
                                     unsigned short* __restrict__ out,
                                     int K, int N) {
  size_t base = (size_t)blockIdx.z * K * N;
  int n = blockIdx.x * 32 + threadIdx.x;
  int k = blockIdx.y * 8 + threadIdx.y;
  if (n < N && k < K)
    out[base + (size_t)n * K + k] = f2bf(in[base + (size_t)k * N + n]);
}

__global__ void embed_kernel(const int* __restrict__ ids,
                             const float* __restrict__ tok,
                             const float* __restrict__ pos,
                             float* __restrict__ x) {
  int row = blockIdx.x;          // b*T + t
  int t = row & 1023;
  int id = ids[row];
  for (int i = threadIdx.x; i < 768; i += 256)
    x[(size_t)row * 768 + i] =
        tok[(size_t)id * 768 + i] + pos[(size_t)t * 768 + i];
}

__global__ __launch_bounds__(256) void layernorm_kernel(
    const float* __restrict__ x, const float* __restrict__ g,
    const float* __restrict__ b, unsigned short* __restrict__ out) {
  __shared__ float red[256];
  __shared__ float mu_s, rstd_s;
  const int tid = threadIdx.x;
  const int row = blockIdx.x;
  const float* xr = x + (size_t)row * 768;
  float v0 = xr[tid], v1 = xr[tid + 256], v2 = xr[tid + 512];
  red[tid] = v0 + v1 + v2;
  __syncthreads();
  for (int o = 128; o > 0; o >>= 1) {
    if (tid < o) red[tid] += red[tid + o];
    __syncthreads();
  }
  if (tid == 0) mu_s = red[0] * (1.f / 768.f);
  __syncthreads();
  float mu = mu_s;
  float d0 = v0 - mu, d1 = v1 - mu, d2 = v2 - mu;
  red[tid] = d0 * d0 + d1 * d1 + d2 * d2;
  __syncthreads();
  for (int o = 128; o > 0; o >>= 1) {
    if (tid < o) red[tid] += red[tid + o];
    __syncthreads();
  }
  if (tid == 0) rstd_s = rsqrtf(red[0] * (1.f / 768.f) + 1e-5f);
  __syncthreads();
  float rstd = rstd_s;
  unsigned short* orow = out + (size_t)row * 768;
  orow[tid]       = f2bf(d0 * rstd * g[tid]       + b[tid]);
  orow[tid + 256] = f2bf(d1 * rstd * g[tid + 256] + b[tid + 256]);
  orow[tid + 512] = f2bf(d2 * rstd * g[tid + 512] + b[tid + 512]);
}

// causal softmax over scores: P[row, s] = softmax(S[row, 0..t]*scale), 0 for s>t
__global__ __launch_bounds__(256) void softmax_causal_kernel(
    const float* __restrict__ S, unsigned short* __restrict__ P, int T) {
  __shared__ float red[256];
  __shared__ float m_s, sum_s;
  const int tid = threadIdx.x;
  const int row = blockIdx.x;          // (b*H+h)*T + t
  const int t = row & (T - 1);
  const float scale = 0.125f;          // 1/sqrt(64)
  const float* sr = S + (size_t)row * T;
  unsigned short* pr = P + (size_t)row * T;
  float mx = -3.4e38f;
  for (int j = tid; j <= t; j += 256) mx = fmaxf(mx, sr[j] * scale);
  red[tid] = mx;
  __syncthreads();
  for (int o = 128; o > 0; o >>= 1) {
    if (tid < o) red[tid] = fmaxf(red[tid], red[tid + o]);
    __syncthreads();
  }
  if (tid == 0) m_s = red[0];
  __syncthreads();
  float m = m_s;
  float s = 0.f;
  for (int j = tid; j <= t; j += 256) s += expf(sr[j] * scale - m);
  red[tid] = s;
  __syncthreads();
  for (int o = 128; o > 0; o >>= 1) {
    if (tid < o) red[tid] += red[tid + o];
    __syncthreads();
  }
  if (tid == 0) sum_s = red[0];
  __syncthreads();
  float inv = 1.f / sum_s;
  for (int j = tid; j < T; j += 256)
    pr[j] = (j <= t) ? f2bf(expf(sr[j] * scale - m) * inv) : (unsigned short)0;
}

__global__ __launch_bounds__(256) void nll_kernel(
    const float* __restrict__ logits, const int* __restrict__ targets,
    float* __restrict__ nll, int Vv) {
  __shared__ float red[256];
  __shared__ float m_s;
  const int tid = threadIdx.x;
  const int row = blockIdx.x;
  const float* lr = logits + (size_t)row * Vv;
  float mx = -3.4e38f;
  for (int j = tid; j < Vv; j += 256) mx = fmaxf(mx, lr[j]);
  red[tid] = mx;
  __syncthreads();
  for (int o = 128; o > 0; o >>= 1) {
    if (tid < o) red[tid] = fmaxf(red[tid], red[tid + o]);
    __syncthreads();
  }
  if (tid == 0) m_s = red[0];
  __syncthreads();
  float m = m_s;
  float s = 0.f;
  for (int j = tid; j < Vv; j += 256) s += expf(lr[j] - m);
  red[tid] = s;
  __syncthreads();
  for (int o = 128; o > 0; o >>= 1) {
    if (tid < o) red[tid] += red[tid + o];
    __syncthreads();
  }
  if (tid == 0) nll[row] = logf(red[0]) + m - lr[targets[row]];
}

__global__ __launch_bounds__(256) void loss_reduce_kernel(
    const float* __restrict__ nll, float* __restrict__ out, int n) {
  __shared__ float red[256];
  const int tid = threadIdx.x;
  float s = 0.f;
  for (int i = tid; i < n; i += 256) s += nll[i];   // fixed order: deterministic
  red[tid] = s;
  __syncthreads();
  for (int o = 128; o > 0; o >>= 1) {
    if (tid < o) red[tid] += red[tid + o];
    __syncthreads();
  }
  if (tid == 0) out[0] = red[0] / (float)n;
}

// ---------------------------------------------------------------------------
// Host orchestration
// ---------------------------------------------------------------------------
extern "C" void kernel_launch(void* const* d_in, const int* in_sizes, int n_in,
                              void* d_out, int out_size, void* d_ws,
                              size_t ws_size, hipStream_t stream) {
  (void)in_sizes; (void)n_in; (void)out_size; (void)ws_size;
  constexpr int Bsz = 4, T = 1024, Dm = 768, Hh = 12, Ll = 6, Kh = 64;
  constexpr int Vv = 32000, F = 3072;
  constexpr int Mtok = Bsz * T;                 // 4096 tokens

  const int*   token_ids = (const int*)  d_in[0];
  const int*   targets   = (const int*)  d_in[1];
  const float* tok_emb   = (const float*)d_in[2];
  const float* pos_emb   = (const float*)d_in[3];
  const float* Wq        = (const float*)d_in[4];
  const float* Wk        = (const float*)d_in[5];
  const float* Wv        = (const float*)d_in[6];
  const float* Wproj     = (const float*)d_in[7];
  const float* bproj     = (const float*)d_in[8];
  const float* ln1_g     = (const float*)d_in[9];
  const float* ln1_b     = (const float*)d_in[10];
  const float* ln2_g     = (const float*)d_in[11];
  const float* ln2_b     = (const float*)d_in[12];
  const float* W1        = (const float*)d_in[13];
  const float* b1        = (const float*)d_in[14];
  const float* W2        = (const float*)d_in[15];
  const float* b2        = (const float*)d_in[16];
  const float* lnf_g     = (const float*)d_in[17];
  const float* lnf_b     = (const float*)d_in[18];
  const float* Whead     = (const float*)d_in[19];
  const float* bhead     = (const float*)d_in[20];
  float* outf = (float*)d_out;                  // logits then loss

  // ---- workspace layout ---------------------------------------------------
  char* wsb = (char*)d_ws;
  size_t off = 0;
  auto take = [&](size_t bytes) -> char* {
    char* p = wsb + off;
    off += (bytes + 255) & ~(size_t)255;
    return p;
  };
  const size_t nWq = (size_t)Ll * Hh * Dm * Kh;
  const size_t nWv = (size_t)Ll * Hh * Dm * Dm;
  const size_t nWp = (size_t)Ll * (Hh * Dm) * Dm;
  const size_t nW1 = (size_t)Ll * Dm * F;
  const size_t nWh = (size_t)Dm * Vv;

  // all weights stored TRANSPOSED (N-major, K-contiguous) in bf16
  unsigned short* WqT  = (unsigned short*)take(nWq * 2);
  unsigned short* WkT  = (unsigned short*)take(nWq * 2);
  unsigned short* WvT  = (unsigned short*)take(nWv * 2);
  unsigned short* WpT  = (unsigned short*)take(nWp * 2);
  unsigned short* W1T  = (unsigned short*)take(nW1 * 2);
  unsigned short* W2T  = (unsigned short*)take(nW1 * 2);
  unsigned short* WhT  = (unsigned short*)take(nWh * 2);
  float*          x    = (float*)         take((size_t)Mtok * Dm * 4);
  unsigned short* h_bf = (unsigned short*)take((size_t)Mtok * Dm * 2);
  unsigned short* q_bf = (unsigned short*)take((size_t)Mtok * Dm * 2);
  unsigned short* k_bf = (unsigned short*)take((size_t)Mtok * Dm * 2);
  unsigned short* v_bf = (unsigned short*)take((size_t)Mtok * Hh * Dm * 2);
  unsigned short* o_bf = (unsigned short*)take((size_t)Mtok * Hh * Dm * 2);
  unsigned short* a1_bf= (unsigned short*)take((size_t)Mtok * F * 2);
  float*          Sbuf = (float*)take((size_t)Bsz * Hh * T * T * 4);
  unsigned short* Pbuf = (unsigned short*)take((size_t)Bsz * Hh * T * T * 2);
  float*          nllbuf = (float*)take((size_t)Mtok * 4);

  // ---- convert + transpose weights to bf16 (N,K) layout -------------------
  const dim3 tb(32, 8);
  transpose_cvt_kernel<<<dim3(2,    96,  Ll*Hh), tb, 0, stream>>>(Wq, WqT, Dm, Kh);
  transpose_cvt_kernel<<<dim3(2,    96,  Ll*Hh), tb, 0, stream>>>(Wk, WkT, Dm, Kh);
  transpose_cvt_kernel<<<dim3(24,   96,  Ll*Hh), tb, 0, stream>>>(Wv, WvT, Dm, Dm);
  transpose_cvt_kernel<<<dim3(24,   1152, Ll),   tb, 0, stream>>>(Wproj, WpT, Hh*Dm, Dm);
  transpose_cvt_kernel<<<dim3(96,   96,  Ll),    tb, 0, stream>>>(W1, W1T, Dm, F);
  transpose_cvt_kernel<<<dim3(24,   384, Ll),    tb, 0, stream>>>(W2, W2T, F, Dm);
  transpose_cvt_kernel<<<dim3(1000, 96,  1),     tb, 0, stream>>>(Whead, WhT, Dm, Vv);

  // ---- embeddings ---------------------------------------------------------
  embed_kernel<<<Mtok, 256, 0, stream>>>(token_ids, tok_emb, pos_emb, x);

  const dim3 blk(256);
  for (int l = 0; l < Ll; ++l) {
    // LN1 -> h_bf
    layernorm_kernel<<<Mtok, blk, 0, stream>>>(x, ln1_g + l * Dm,
                                               ln1_b + l * Dm, h_bf);
    // q, k, v projections (per head); weights are (N,K) so TB path
    for (int hh = 0; hh < Hh; ++hh) {
      const size_t wqk = ((size_t)(l * Hh + hh)) * Dm * Kh;
      gemm_bf16_wmma<true, false, false, false, true>
          <<<dim3(1, Mtok / 128), blk, 0, stream>>>(
              h_bf, Dm, WqT + wqk, Dm, nullptr, nullptr,
              q_bf + hh * Kh, Dm, Mtok, Kh, Dm);
      gemm_bf16_wmma<true, false, false, false, true>
          <<<dim3(1, Mtok / 128), blk, 0, stream>>>(
              h_bf, Dm, WkT + wqk, Dm, nullptr, nullptr,
              k_bf + hh * Kh, Dm, Mtok, Kh, Dm);
      const size_t wv = ((size_t)(l * Hh + hh)) * Dm * Dm;
      gemm_bf16_wmma<true, false, false, false, true>
          <<<dim3(Dm / 128, Mtok / 128), blk, 0, stream>>>(
              h_bf, Dm, WvT + wv, Dm, nullptr, nullptr,
              v_bf + (size_t)hh * Dm, Hh * Dm, Mtok, Dm, Dm);
    }
    // scores S = q @ k^T per (b,h)   (k is naturally (N=s, K) -> TB)
    for (int b = 0; b < Bsz; ++b)
      for (int hh = 0; hh < Hh; ++hh) {
        const size_t qk = (size_t)b * T * Dm + (size_t)hh * Kh;
        gemm_bf16_wmma<true, false, false, false, false>
            <<<dim3(T / 128, T / 128), blk, 0, stream>>>(
                q_bf + qk, Dm, k_bf + qk, Dm, nullptr, nullptr,
                Sbuf + ((size_t)(b * Hh + hh)) * T * T, T, T, T, Kh);
      }
    // causal softmax -> P (bf16, zero above diagonal)
    softmax_causal_kernel<<<Bsz * Hh * T, blk, 0, stream>>>(Sbuf, Pbuf, T);
    // o = P @ v per (b,h)   (v is (K=s, N=e) -> non-TB path)
    for (int b = 0; b < Bsz; ++b)
      for (int hh = 0; hh < Hh; ++hh) {
        const size_t vo = (size_t)b * T * Hh * Dm + (size_t)hh * Dm;
        gemm_bf16_wmma<false, false, false, false, true>
            <<<dim3(Dm / 128, T / 128), blk, 0, stream>>>(
                Pbuf + ((size_t)(b * Hh + hh)) * T * T, T,
                v_bf + vo, Hh * Dm, nullptr, nullptr,
                o_bf + vo, Hh * Dm, T, Dm, T);
      }
    // x += o @ Wproj + bproj
    gemm_bf16_wmma<true, true, true, false, false>
        <<<dim3(Dm / 128, Mtok / 128), blk, 0, stream>>>(
            o_bf, Hh * Dm, WpT + (size_t)l * Hh * Dm * Dm, Hh * Dm,
            bproj + l * Dm, x, x, Dm, Mtok, Dm, Hh * Dm);
    // LN2 -> h_bf
    layernorm_kernel<<<Mtok, blk, 0, stream>>>(x, ln2_g + l * Dm,
                                               ln2_b + l * Dm, h_bf);
    // a1 = relu(h @ W1 + b1)
    gemm_bf16_wmma<true, true, false, true, true>
        <<<dim3(F / 128, Mtok / 128), blk, 0, stream>>>(
            h_bf, Dm, W1T + (size_t)l * Dm * F, Dm,
            b1 + l * F, nullptr, a1_bf, F, Mtok, F, Dm);
    // x += a1 @ W2 + b2
    gemm_bf16_wmma<true, true, true, false, false>
        <<<dim3(Dm / 128, Mtok / 128), blk, 0, stream>>>(
            a1_bf, F, W2T + (size_t)l * F * Dm, F,
            b2 + l * Dm, x, x, Dm, Mtok, Dm, F);
  }

  // final LN -> h_bf; logits = h @ Whead + bhead (f32 into d_out)
  layernorm_kernel<<<Mtok, blk, 0, stream>>>(x, lnf_g, lnf_b, h_bf);
  gemm_bf16_wmma<true, true, false, false, false>
      <<<dim3(Vv / 128, Mtok / 128), blk, 0, stream>>>(
          h_bf, Dm, WhT, Dm, bhead, nullptr, outf, Vv, Mtok, Vv, Dm);

  // loss = mean over rows of (logsumexp - logit[target])
  nll_kernel<<<Mtok, blk, 0, stream>>>(outf, targets, nllbuf, Vv);
  loss_reduce_kernel<<<1, blk, 0, stream>>>(nllbuf,
                                            outf + (size_t)Mtok * Vv, Mtok);
}
